// GlobalCapsulePooling_2_32392643346843
// MI455X (gfx1250) — compile-verified, matrix operand order checked by reading, not run on hardware
//
#include <hip/hip_runtime.h>
#include <hip/hip_bf16.h>

typedef _Float16 half16 __attribute__((ext_vector_type(16)));
typedef _Float16 half8  __attribute__((ext_vector_type(8)));
typedef _Float16 half4v __attribute__((ext_vector_type(4)));
typedef float    float8 __attribute__((ext_vector_type(8)));

static constexpr int N_NODES = 51200;
static constexpr int N_EDGES = 614400;
static constexpr int NG = 256;   // graphs
static constexpr int NT = 8;     // capsule heads
static constexpr int ND = 128;   // channels
static constexpr int TD = NT * ND;

__device__ __forceinline__ void atomAddF(float* p, float v) {
    (void)__hip_atomic_fetch_add(p, v, __ATOMIC_RELAXED, __HIP_MEMORY_SCOPE_AGENT);
}

// ---------------- utility ----------------
__global__ void fill_f32(float* p, float v, int n) {
    int i = blockIdx.x * 256 + threadIdx.x;
    if (i < n) p[i] = v;
}

// ---------------- BatchNorm ----------------
__global__ void bn_partial(const float* __restrict__ x, float* __restrict__ sums) {
    __shared__ float sh[512];
    int tid = threadIdx.x;
    int d = tid & 127;
    int rsub = tid >> 7;               // 0/1 : two rows in flight per block
    float s = 0.f, q = 0.f;
    for (int n = blockIdx.x * 2 + rsub; n < N_NODES; n += gridDim.x * 2) {
        float v = x[(size_t)n * ND + d];
        s += v; q += v * v;
    }
    sh[tid] = s; sh[256 + tid] = q;
    __syncthreads();
    if (tid < 128) {
        atomAddF(&sums[d],        sh[tid] + sh[tid + 128]);
        atomAddF(&sums[128 + d],  sh[256 + tid] + sh[256 + tid + 128]);
    }
}

__global__ void bn_final(const float* __restrict__ sums,
                         const float* __restrict__ gamma,
                         const float* __restrict__ beta,
                         float* __restrict__ ss) {
    int d = threadIdx.x;               // 128 threads
    float mu  = sums[d] * (1.0f / N_NODES);
    float var = sums[128 + d] * (1.0f / N_NODES) - mu * mu;
    float sc  = gamma[d] * rsqrtf(var + 1e-5f);
    ss[d]       = sc;
    ss[128 + d] = beta[d] - mu * sc;
}

__global__ void normalize_k(const float* __restrict__ x, const float* __restrict__ ss,
                            float* __restrict__ xn, _Float16* __restrict__ xh) {
    int idx = blockIdx.x * 256 + threadIdx.x;     // N*D threads
    int d = idx & 127;
    float v = x[idx] * ss[d] + ss[128 + d];
    xn[idx] = v;
    xh[idx] = (_Float16)v;
}

// ---------------- W pre-pack into WMMA B-fragment lane layout ----------------
// linear idx = (((t*8 + colTile)*4 + kChunk)*32 + lane)*16 + e
// lane: lo=lane&15 -> N column, hi=lane>>4 ; element e -> K per ISA 16-bit layout
__global__ void wpack_k(const float* __restrict__ W, _Float16* __restrict__ wp) {
    int idx = blockIdx.x * 256 + threadIdx.x;     // T*D*D threads
    int e    =  idx        & 15;
    int lane = (idx >> 4)  & 31;
    int kc   = (idx >> 9)  & 3;
    int ct   = (idx >> 11) & 7;
    int t    =  idx >> 14;
    int lo = lane & 15, hi = lane >> 4;
    int k   = kc * 32 + ((e & 8) ? 16 : 0) + (e & 7) + hi * 8;
    int col = ct * 16 + lo;
    wp[idx] = (_Float16)W[(size_t)t * ND * ND + (size_t)k * ND + col];
}

// ---------------- degree / gcn_norm ----------------
__global__ void deg_acc(const int* __restrict__ ei, const float* __restrict__ ew,
                        float* __restrict__ deg, float* __restrict__ ndeg) {
    int e = blockIdx.x * 256 + threadIdx.x;
    if (e >= N_EDGES) return;
    float w = ew[e];
    atomAddF(&deg[ei[N_EDGES + e]], w);   // col  (deg pre-initialized to 1.0 = self loop)
    atomAddF(&ndeg[ei[e]], w);            // row  (node_deg, edges only)
}

__global__ void dis_k(const float* __restrict__ deg, float* __restrict__ dis) {
    int n = blockIdx.x * 256 + threadIdx.x;
    if (n >= N_NODES) return;
    float d = deg[n];
    dis[n] = d > 0.f ? rsqrtf(fmaxf(d, 1e-12f)) : 0.f;
}

// ---------------- WMMA GEMM: u_hat = bias + dis^2 * (xn @ W[t]), xWh = f16(xn@W[t]) ----------------
// grid = (N/16, T), block = 256 (8 waves); wave w owns column tile w (16 cols), 16x16 C tile.
__global__ void gemm_uhat(const _Float16* __restrict__ xh,
                          const _Float16* __restrict__ wpack,
                          const float* __restrict__ bias,
                          const float* __restrict__ dis,
                          float* __restrict__ u_hat,
                          _Float16* __restrict__ xWh) {
    const int wave = threadIdx.x >> 5;        // 0..7 == column tile
    const int lane = threadIdx.x & 31;
    const int lo = lane & 15, hi = lane >> 4;
    const int rowBase = blockIdx.x * 16;
    const int t = blockIdx.y;
    const int colBase = wave * 16;

    float8 acc = {};
    const int arow = rowBase + lo;
    const _Float16* parow = xh + (size_t)arow * ND + hi * 8;
    const _Float16* pbbase = wpack + (size_t)(((t * 8 + wave) * 4) * 32 + lane) * 16;
#pragma unroll
    for (int kc = 0; kc < 4; ++kc) {
        const _Float16* pa = parow + kc * 32;
        half8 a0 = *(const half8*)(pa);
        half8 a1 = *(const half8*)(pa + 16);
        half16 a;
#pragma unroll
        for (int i = 0; i < 8; ++i) { a[i] = a0[i]; a[i + 8] = a1[i]; }
        const _Float16* pb = pbbase + (size_t)kc * (32 * 16);
        half8 b0 = *(const half8*)(pb);
        half8 b1 = *(const half8*)(pb + 8);
        half16 b;
#pragma unroll
        for (int i = 0; i < 8; ++i) { b[i] = b0[i]; b[i + 8] = b1[i]; }
        acc = __builtin_amdgcn_wmma_f32_16x16x32_f16(false, a, false, b,
                                                     (short)0, acc, false, false);
    }

    const int d = colBase + lo;
    const float bt = bias[t * ND + d];
#pragma unroll
    for (int r = 0; r < 8; ++r) {
        int n = rowBase + r + hi * 8;
        float v = acc[r];
        size_t idx = (size_t)n * TD + t * ND + d;
        xWh[idx] = (_Float16)v;
        float ds = dis[n];
        u_hat[idx] = bt + ds * ds * v;      // fused self-loop term
    }
}

// ---------------- edge message scatter: u_hat[col] += norm * xW[row] ----------------
// one wave per edge; lane handles 4 contiguous d's; loop over t.
__global__ void edge_msg(const int* __restrict__ ei, const float* __restrict__ ew,
                         const float* __restrict__ dis,
                         const _Float16* __restrict__ xWh, float* __restrict__ u_hat) {
    int wave = threadIdx.x >> 5, lane = threadIdx.x & 31;
    int e = blockIdx.x * 8 + wave;           // E divisible by 8
    int r = ei[e], c = ei[N_EDGES + e];
    float nrm = dis[r] * ew[e] * dis[c];
    int d0 = lane * 4;
    size_t rb = (size_t)r * TD + d0;
    size_t cb = (size_t)c * TD + d0;
#pragma unroll
    for (int t = 0; t < NT; ++t) {
        half4v m = *(const half4v*)(xWh + rb + (size_t)t * ND);
#pragma unroll
        for (int j = 0; j < 4; ++j)
            atomAddF(&u_hat[cb + (size_t)t * ND + j], nrm * (float)m[j]);
    }
}

// ---------------- per-graph mean of normalized x ----------------
__global__ void xmean_acc(const float* __restrict__ xn, const int* __restrict__ batch,
                          float* __restrict__ x_mean) {
    int idx = blockIdx.x * 256 + threadIdx.x;   // N*D
    int n = idx >> 7, d = idx & 127;
    atomAddF(&x_mean[(size_t)batch[n] * ND + d], xn[idx]);
}

__global__ void cnt_acc(const int* __restrict__ batch, float* __restrict__ cnt) {
    int n = blockIdx.x * 256 + threadIdx.x;
    if (n >= N_NODES) return;
    atomAddF(&cnt[batch[n]], 1.0f);
}

__global__ void xmean_fin(const float* __restrict__ cnt, float* __restrict__ x_mean) {
    int idx = blockIdx.x * 256 + threadIdx.x;   // G*D
    int g = idx >> 7;
    x_mean[idx] /= fmaxf(cnt[g], 1.0f);
}

// ---------------- dynamic routing ----------------
__global__ void b_init(const float* __restrict__ ndeg, float* __restrict__ b) {
    int idx = blockIdx.x * 256 + threadIdx.x;   // N*T
    if (idx >= N_NODES * NT) return;
    b[idx] = ndeg[idx >> 3];
}

// block = node: softmax over T in regs, scatter c*u_hat into s[g]
__global__ void s_acc(const float* __restrict__ b, const float* __restrict__ u_hat,
                      const int* __restrict__ batch, float* __restrict__ s) {
    int n = blockIdx.x;
    int tid = threadIdx.x;
    int t = tid >> 5, lane = tid & 31;
    float bv[NT];
    float mx = -1e30f;
#pragma unroll
    for (int i = 0; i < NT; ++i) { bv[i] = b[n * NT + i]; mx = fmaxf(mx, bv[i]); }
    float den = 0.f;
#pragma unroll
    for (int i = 0; i < NT; ++i) { bv[i] = __expf(bv[i] - mx); den += bv[i]; }
    float ct = bv[t] / den;
    int g = batch[n];
    int d0 = lane * 4;
    size_t su = (size_t)n * TD + t * ND + d0;
    size_t sg = (size_t)g * TD + t * ND + d0;
#pragma unroll
    for (int j = 0; j < 4; ++j)
        atomAddF(&s[sg + j], ct * u_hat[su + j]);
}

// squash s -> v ; final round adds x_mean and emits per-capsule L2 norm
__global__ void squash_k(const float* __restrict__ s, const float* __restrict__ x_mean,
                         float* __restrict__ v, float* __restrict__ out, int final_) {
    int g = blockIdx.x >> 3, t = blockIdx.x & 7;
    int d = threadIdx.x;                        // 128 threads
    size_t idx = (size_t)blockIdx.x * ND + d;
    float sv = s[idx];
    if (final_) sv += x_mean[(size_t)g * ND + d];
    __shared__ float sh[128];
    sh[d] = sv * sv;
    __syncthreads();
    for (int off = 64; off > 0; off >>= 1) {
        if (d < off) sh[d] += sh[d + off];
        __syncthreads();
    }
    float s2 = sh[0];
    float k = (s2 / (1.f + s2)) * rsqrtf(s2 + 1e-16f);
    if (final_) {
        if (d == 0) out[g * NT + t] = sqrtf(k * k * s2 + 1e-16f);
    } else {
        v[idx] = k * sv;
    }
}

// b += dot(u_hat[n,t,:], v[g,t,:]) ; block = node, 32 lanes per t
__global__ void b_upd(const float* __restrict__ u_hat, const float* __restrict__ v,
                      const int* __restrict__ batch, float* __restrict__ b) {
    int n = blockIdx.x;
    int tid = threadIdx.x;
    int t = tid >> 5, lane = tid & 31;
    int g = batch[n];
    int d0 = lane * 4;
    size_t iu = (size_t)n * TD + t * ND + d0;
    size_t iv = (size_t)g * TD + t * ND + d0;
    float p = 0.f;
#pragma unroll
    for (int j = 0; j < 4; ++j) p += u_hat[iu + j] * v[iv + j];
    __shared__ float sh[256];
    sh[tid] = p;
    __syncthreads();
    for (int off = 16; off > 0; off >>= 1) {
        if (lane < off) sh[tid] += sh[tid + off];
        __syncthreads();
    }
    if (lane == 0) b[n * NT + t] += sh[tid];
}

extern "C" void kernel_launch(void* const* d_in, const int* in_sizes, int n_in,
                              void* d_out, int out_size, void* d_ws, size_t ws_size,
                              hipStream_t stream) {
    const float* x     = (const float*)d_in[0];
    const float* ew    = (const float*)d_in[1];
    const float* gamma = (const float*)d_in[2];
    const float* beta  = (const float*)d_in[3];
    const float* W     = (const float*)d_in[4];
    const float* bias  = (const float*)d_in[5];
    const int*   ei    = (const int*)d_in[6];
    const int*   batch = (const int*)d_in[7];
    float* out = (float*)d_out;

    char* base = (char*)d_ws;
    size_t off = 0;
    auto carve = [&](size_t bytes) -> char* {
        char* p = base + off;
        off += (bytes + 255) & ~(size_t)255;
        return p;
    };

    float*    bn_sums = (float*)   carve(256 * sizeof(float));
    float*    ss      = (float*)   carve(256 * sizeof(float));
    float*    xn      = (float*)   carve((size_t)N_NODES * ND * sizeof(float));
    _Float16* xh      = (_Float16*)carve((size_t)N_NODES * ND * sizeof(_Float16));
    _Float16* wpack   = (_Float16*)carve((size_t)NT * ND * ND * sizeof(_Float16));
    _Float16* xWh     = (_Float16*)carve((size_t)N_NODES * TD * sizeof(_Float16));
    float*    u_hat   = (float*)   carve((size_t)N_NODES * TD * sizeof(float));
    float*    deg     = (float*)   carve((size_t)N_NODES * sizeof(float));
    float*    ndeg    = (float*)   carve((size_t)N_NODES * sizeof(float));
    float*    dis     = (float*)   carve((size_t)N_NODES * sizeof(float));
    float*    bvec    = (float*)   carve((size_t)N_NODES * NT * sizeof(float));
    float*    s       = (float*)   carve((size_t)NG * TD * sizeof(float));
    float*    vv      = (float*)   carve((size_t)NG * TD * sizeof(float));
    float*    x_mean  = (float*)   carve((size_t)NG * ND * sizeof(float));
    float*    cnt     = (float*)   carve((size_t)NG * sizeof(float));

    // ---- init ----
    fill_f32<<<1, 256, 0, stream>>>(bn_sums, 0.f, 256);
    fill_f32<<<(N_NODES + 255) / 256, 256, 0, stream>>>(deg, 1.0f, N_NODES);   // self loop
    fill_f32<<<(N_NODES + 255) / 256, 256, 0, stream>>>(ndeg, 0.f, N_NODES);
    fill_f32<<<(NG * ND + 255) / 256, 256, 0, stream>>>(x_mean, 0.f, NG * ND);
    fill_f32<<<1, 256, 0, stream>>>(cnt, 0.f, NG);

    // ---- BatchNorm ----
    bn_partial<<<256, 256, 0, stream>>>(x, bn_sums);
    bn_final<<<1, 128, 0, stream>>>(bn_sums, gamma, beta, ss);
    normalize_k<<<(N_NODES * ND) / 256, 256, 0, stream>>>(x, ss, xn, xh);

    // ---- GCN norm prep ----
    wpack_k<<<(NT * ND * ND) / 256, 256, 0, stream>>>(W, wpack);
    deg_acc<<<(N_EDGES + 255) / 256, 256, 0, stream>>>(ei, ew, deg, ndeg);
    dis_k<<<(N_NODES + 255) / 256, 256, 0, stream>>>(deg, dis);

    // ---- WMMA matmul + fused self-loop epilogue ----
    dim3 ggrid(N_NODES / 16, NT);
    gemm_uhat<<<ggrid, 256, 0, stream>>>(xh, wpack, bias, dis, u_hat, xWh);

    // ---- edge scatter (dominant traffic; u_hat ~L2-resident) ----
    edge_msg<<<N_EDGES / 8, 256, 0, stream>>>(ei, ew, dis, xWh, u_hat);

    // ---- per-graph mean ----
    xmean_acc<<<(N_NODES * ND) / 256, 256, 0, stream>>>(xn, batch, x_mean);
    cnt_acc<<<(N_NODES + 255) / 256, 256, 0, stream>>>(batch, cnt);
    xmean_fin<<<(NG * ND) / 256, 256, 0, stream>>>(cnt, x_mean);

    // ---- dynamic routing ----
    b_init<<<(N_NODES * NT + 255) / 256, 256, 0, stream>>>(ndeg, bvec);
    for (int it = 0; it < 3; ++it) {
        fill_f32<<<(NG * TD + 255) / 256, 256, 0, stream>>>(s, 0.f, NG * TD);
        s_acc<<<N_NODES, 256, 0, stream>>>(bvec, u_hat, batch, s);
        if (it < 2) {
            squash_k<<<NG * NT, 128, 0, stream>>>(s, x_mean, vv, out, 0);
            b_upd<<<N_NODES, 256, 0, stream>>>(u_hat, vv, batch, bvec);
        } else {
            squash_k<<<NG * NT, 128, 0, stream>>>(s, x_mean, vv, out, 1);
        }
    }
}
